// Ximp_5557687681836
// MI455X (gfx1250) — compile-verified, matrix operand order checked by reading, not run on hardware
//
#include <hip/hip_runtime.h>
#include <hip/hip_bf16.h>

typedef __attribute__((ext_vector_type(16))) _Float16 v16h;
typedef __attribute__((ext_vector_type(8)))  float    v8f;
typedef __attribute__((ext_vector_type(4)))  float    v4f;

#define H_DIM 128
#define BN_EPS 1e-5f

// ---------------------------------------------------------------------------
// Generic fused WMMA GEMM:  C[M,N] (+)= f_out( f_in(A [+Aadd]) @ W )
//   in_mode:  0 = plain, 1 = BN(mean,inv)+ReLU on A, 2 = ReLU on A
//   out_mode: 0 = store, 1 = C += relu(acc), 2 = C += acc
// Block = 256 threads = 8 waves; block tile = 128 rows x 64 cols,
// grid.y slices N in 64-column chunks.
// Weights staged in LDS as f16, PRE-SWIZZLED into per-lane fragment order:
//   Ws[(((k0/32)*4 + t)*32 + lane)*16 + j] = W[k(k0,lane,j)][colBase + t*16 + (lane&15)]
// so each lane's B fragment is one contiguous 32-byte v16h load (2x ds_load_b128).
// ---------------------------------------------------------------------------
__global__ void gemm_wmma_k(const float* __restrict__ A, const float* __restrict__ Aadd,
                            const float* __restrict__ mean, const float* __restrict__ inv,
                            const float* __restrict__ W, float* __restrict__ C,
                            int Mrows, int K, int N, int in_mode, int out_mode) {
    __shared__ __attribute__((aligned(32))) _Float16 Ws[256 * 64]; // up to 32 KB
    __shared__ float mS[256];
    __shared__ float iS[256];

    const int tid = threadIdx.x;
    const int colBase = blockIdx.y * 64;

    // stage weight slice into LDS as f16, swizzled to per-lane fragment order
    for (int idx = tid; idx < K * 64; idx += 256) {
        int j  = idx & 15;
        int l  = (idx >> 4) & 31;
        int t  = (idx >> 9) & 3;
        int kb = idx >> 11;
        int kh = (l >> 4) << 3;
        int k  = kb * 32 + ((j < 8) ? (kh + j) : (16 + kh + (j - 8)));
        int n  = colBase + t * 16 + (l & 15);
        Ws[idx] = (_Float16)W[(size_t)k * N + n];
    }
    if (in_mode == 1) {
        for (int k = tid; k < K; k += 256) { mS[k] = mean[k]; iS[k] = inv[k]; }
    }
    __syncthreads();

    const int wave = tid >> 5;
    const int lane = tid & 31;
    const int rowBase = blockIdx.x * 128 + wave * 16;
    const int m     = lane & 15;
    const int khalf = (lane >> 4) << 3;      // 0 or 8 (ISA A/B K-split)
    const int row   = rowBase + m;
    const bool rowOK = row < Mrows;

    v8f acc[4];
    acc[0] = {}; acc[1] = {}; acc[2] = {}; acc[3] = {};

    for (int k0 = 0; k0 < K; k0 += 32) {
        // ---- A fragment: two contiguous 8-float runs, 32B-aligned -> b128 loads
        float av[16];
        if (rowOK) {
            const float* base = &A[(size_t)row * K + k0];
            const v4f* p0 = (const v4f*)(base + khalf);
            const v4f* p1 = (const v4f*)(base + 16 + khalf);
            v4f c0 = p0[0], c1 = p0[1], c2 = p1[0], c3 = p1[1];
            if (Aadd) {
                const float* ab = &Aadd[(size_t)row * K + k0];
                const v4f* q0 = (const v4f*)(ab + khalf);
                const v4f* q1 = (const v4f*)(ab + 16 + khalf);
                c0 += q0[0]; c1 += q0[1]; c2 += q1[0]; c3 += q1[1];
            }
#pragma unroll
            for (int j = 0; j < 4; ++j) {
                av[j] = c0[j]; av[4 + j] = c1[j]; av[8 + j] = c2[j]; av[12 + j] = c3[j];
            }
        } else {
#pragma unroll
            for (int j = 0; j < 16; ++j) av[j] = 0.0f;
        }
        v16h a;
#pragma unroll
        for (int j = 0; j < 16; ++j) {
            int k = k0 + ((j < 8) ? (khalf + j) : (16 + khalf + (j - 8)));
            float v = av[j];
            if (in_mode == 1)      v = fmaxf((v - mS[k]) * iS[k], 0.0f);
            else if (in_mode == 2) v = fmaxf(v, 0.0f);
            a[j] = (_Float16)v;
        }
        // ---- 4 column tiles; B fragment = one contiguous v16h LDS load
#pragma unroll
        for (int t = 0; t < 4; ++t) {
            int bidx = (((k0 >> 5) * 4 + t) * 32 + lane) * 16;
            v16h b = *(const v16h*)&Ws[bidx];
            acc[t] = __builtin_amdgcn_wmma_f32_16x16x32_f16(
                false, a, false, b, (short)0, acc[t], false, false);
        }
    }

#pragma unroll
    for (int t = 0; t < 4; ++t) {
#pragma unroll
        for (int r = 0; r < 8; ++r) {
            int orow = rowBase + r + ((lane >> 4) << 3);
            int ocol = colBase + t * 16 + (lane & 15);
            if (orow < Mrows) {
                float v = acc[t][r];
                float* p = &C[(size_t)orow * N + ocol];
                if (out_mode == 0)      *p = v;
                else if (out_mode == 1) *p += fmaxf(v, 0.0f);
                else                    *p += v;
            }
        }
    }
}

// ---------------------------------------------------------------------------
// Elementwise / reduction helpers
// ---------------------------------------------------------------------------
__global__ void zero_f32(float* __restrict__ p, int n) {
    for (int i = blockIdx.x * blockDim.x + threadIdx.x; i < n; i += gridDim.x * blockDim.x)
        p[i] = 0.0f;
}

__global__ void colstats(const float* __restrict__ T, int Mrows, int N,
                         float* __restrict__ sum, float* __restrict__ ssq) {
    int c = threadIdx.x;
    float s = 0.0f, q = 0.0f;
    for (int r = blockIdx.x; r < Mrows; r += gridDim.x) {
        float v = T[(size_t)r * N + c];
        s += v; q += v * v;
    }
    atomicAdd(&sum[c], s);
    atomicAdd(&ssq[c], q);
}

__global__ void finstats(const float* __restrict__ sum, const float* __restrict__ ssq,
                         int Mrows, int N, float* __restrict__ mean, float* __restrict__ inv) {
    int c = blockIdx.x * blockDim.x + threadIdx.x;
    if (c < N) {
        float mu  = sum[c] / (float)Mrows;
        float var = ssq[c] / (float)Mrows - mu * mu;
        mean[c] = mu;
        inv[c]  = rsqrtf(var + BN_EPS);
    }
}

__global__ void norm_relu(const float* __restrict__ B, const float* __restrict__ mean,
                          const float* __restrict__ inv, float* __restrict__ X, int total, int N) {
    for (int i = blockIdx.x * blockDim.x + threadIdx.x; i < total; i += gridDim.x * blockDim.x) {
        int c = i & (N - 1);
        X[i] = fmaxf((B[i] - mean[c]) * inv[c], 0.0f);
    }
}

// ---------------------------------------------------------------------------
// Encoders / graph ops
// ---------------------------------------------------------------------------
__global__ void atom_enc(const int* __restrict__ node_feat, const float* __restrict__ atom_emb,
                         float* __restrict__ X, int Nn) {
    int n = blockIdx.x, h = threadIdx.x;
    if (n >= Nn) return;
    float s = 0.0f;
#pragma unroll
    for (int f = 0; f < 9; ++f) {
        int v = node_feat[n * 9 + f];
        s += atom_emb[((size_t)(f * 100 + v)) * H_DIM + h];
    }
    X[(size_t)n * H_DIM + h] = s;
}

__global__ void rg_init(const int* __restrict__ feat, const float* __restrict__ emb,
                        float* __restrict__ RG, int Nr) {
    int n = blockIdx.x, h = threadIdx.x;
    if (n >= Nr) return;
    RG[(size_t)n * H_DIM + h] = emb[(size_t)feat[n] * H_DIM + h];
}

__global__ void edge_agg(const int* __restrict__ ei, const int* __restrict__ ef,
                         const float* __restrict__ bemb, const float* __restrict__ X,
                         float* __restrict__ AGG, int Ecnt) {
    int e = blockIdx.x, h = threadIdx.x;
    if (e >= Ecnt) return;
    int s = ei[e], d = ei[Ecnt + e];
    float v = X[(size_t)s * H_DIM + h];
#pragma unroll
    for (int j = 0; j < 3; ++j) {
        int f = ef[e * 3 + j];
        v += bemb[((size_t)(j * 100 + f)) * H_DIM + h];
    }
    v = fmaxf(v, 0.0f);
    atomicAdd(&AGG[(size_t)d * H_DIM + h], v);
}

__global__ void rg_agg(const int* __restrict__ rei, const float* __restrict__ RG,
                       float* __restrict__ AGG, int Ecnt) {
    int e = blockIdx.x, h = threadIdx.x;
    if (e >= Ecnt) return;
    int s = rei[e], d = rei[Ecnt + e];
    atomicAdd(&AGG[(size_t)d * H_DIM + h], RG[(size_t)s * H_DIM + h]);
}

__global__ void scat_add(const int* __restrict__ src_idx, const int* __restrict__ dst_idx,
                         const float* __restrict__ SRC, float* __restrict__ DST,
                         float* __restrict__ cnt, int Mm) {
    int m = blockIdx.x, h = threadIdx.x;
    if (m >= Mm) return;
    int r = src_idx[m], c = dst_idx[m];
    atomicAdd(&DST[(size_t)c * H_DIM + h], SRC[(size_t)r * H_DIM + h]);
    if (h == 0) atomicAdd(&cnt[c], 1.0f);
}

__global__ void pool_add(const int* __restrict__ bidx, const float* __restrict__ X,
                         float* __restrict__ P, float* __restrict__ cnt, int rows) {
    int n = blockIdx.x, h = threadIdx.x;
    if (n >= rows) return;
    int g = bidx[n];
    atomicAdd(&P[(size_t)g * H_DIM + h], X[(size_t)n * H_DIM + h]);
    if (h == 0) atomicAdd(&cnt[g], 1.0f);
}

__global__ void div_cnt(float* __restrict__ DST, const float* __restrict__ cnt, int rows) {
    int r = blockIdx.x, h = threadIdx.x;
    if (r >= rows) return;
    float c = fmaxf(cnt[r], 1.0f);
    DST[(size_t)r * H_DIM + h] /= c;
}

// ---------------------------------------------------------------------------
extern "C" void kernel_launch(void* const* d_in, const int* in_sizes, int n_in,
                              void* d_out, int out_size, void* d_ws, size_t ws_size,
                              hipStream_t stream) {
    const int H = H_DIM, L = 3;
    const int* node_feat   = (const int*)d_in[0];
    const int* edge_index  = (const int*)d_in[1];
    const int* edge_feat   = (const int*)d_in[2];
    const int* batch       = (const int*)d_in[3];
    const int* mapping_row = (const int*)d_in[4];
    const int* mapping_col = (const int*)d_in[5];
    const int* rg_edge_idx = (const int*)d_in[6];
    const int* rg_feat     = (const int*)d_in[7];
    const int* tree_batch  = (const int*)d_in[8];
    const float* atom_emb   = (const float*)d_in[10];
    const float* bond_emb   = (const float*)d_in[11];
    const float* rg_emb     = (const float*)d_in[12];
    const float* atom_W1    = (const float*)d_in[13];
    const float* atom_W2    = (const float*)d_in[14];
    const float* rg_W1      = (const float*)d_in[15];
    const float* rg_W2      = (const float*)d_in[16];
    const float* raw2rg_W   = (const float*)d_in[17];
    const float* rg2raw_W   = (const float*)d_in[18];
    const float* atom_lin_W = (const float*)d_in[19];
    const float* rg_lin_W   = (const float*)d_in[20];
    const float* lin_W      = (const float*)d_in[21];

    const int N   = in_sizes[0] / 9;
    const int E   = in_sizes[1] / 2;
    const int Mm  = in_sizes[4];
    const int ERG = in_sizes[6] / 2;
    const int NRG = in_sizes[7];
    const int G   = out_size / 128;   // OUT == 128

    float* ws = (float*)d_ws;
    float* X    = ws;  ws += (size_t)N * H;
    float* RG   = ws;  ws += (size_t)NRG * H;
    float* AGG  = ws;  ws += (size_t)N * H;
    float* T    = ws;  ws += (size_t)N * 2 * H;   // GEMM1 out; reused as scatter buffer
    float* B2   = ws;  ws += (size_t)N * H;       // GEMM2 raw out
    float* CNT  = ws;  ws += N;
    float* SUM  = ws;  ws += 256;
    float* SSQ  = ws;  ws += 256;
    float* MEAN = ws;  ws += 256;
    float* INV  = ws;  ws += 256;
    float* PX   = ws;  ws += (size_t)G * H;
    float* PRG  = ws;  ws += (size_t)G * H;
    float* Z    = ws;  ws += (size_t)G * H;

    auto gemm = [&](const float* A, const float* Aadd, const float* mean, const float* inv,
                    const float* W, float* C, int Mr, int K, int Nc, int im, int om) {
        dim3 grid((Mr + 127) / 128, Nc / 64);
        gemm_wmma_k<<<grid, 256, 0, stream>>>(A, Aadd, mean, inv, W, C, Mr, K, Nc, im, om);
    };
    auto bnstats = [&](const float* Tp, int Mr, int Nc) {
        zero_f32<<<2, 256, 0, stream>>>(SUM, 512);   // SUM+SSQ contiguous
        colstats<<<512, Nc, 0, stream>>>(Tp, Mr, Nc, SUM, SSQ);
        finstats<<<1, Nc, 0, stream>>>(SUM, SSQ, Mr, Nc, MEAN, INV);
    };

    // encoders
    atom_enc<<<N, 128, 0, stream>>>(node_feat, atom_emb, X, N);
    rg_init<<<NRG, 128, 0, stream>>>(rg_feat, rg_emb, RG, NRG);

    for (int i = 0; i < L; ++i) {
        // GINE edge aggregation
        zero_f32<<<2048, 256, 0, stream>>>(AGG, N * H);
        edge_agg<<<E, 128, 0, stream>>>(edge_index, edge_feat,
                                        bond_emb + (size_t)i * 3 * 100 * H, X, AGG, E);
        // atom MLP: T = (X+AGG)@W1 ; B2 = relu(BN(T))@W2 ; X = relu(BN(B2))
        gemm(X, AGG, nullptr, nullptr, atom_W1 + (size_t)i * H * 2 * H, T, N, H, 2 * H, 0, 0);
        bnstats(T, N, 2 * H);
        gemm(T, nullptr, MEAN, INV, atom_W2 + (size_t)i * 2 * H * H, B2, N, 2 * H, H, 1, 0);
        bnstats(B2, N, H);
        norm_relu<<<2048, 256, 0, stream>>>(B2, MEAN, INV, X, N * H, H);

        // raw -> reduced-graph: RG += relu(scatter_mean(X[row]->col) @ raw2rg_W)
        zero_f32<<<2048, 256, 0, stream>>>(T, NRG * H);
        zero_f32<<<256, 256, 0, stream>>>(CNT, NRG);
        scat_add<<<Mm, 128, 0, stream>>>(mapping_row, mapping_col, X, T, CNT, Mm);
        div_cnt<<<NRG, 128, 0, stream>>>(T, CNT, NRG);
        gemm(T, nullptr, nullptr, nullptr, raw2rg_W + (size_t)i * H * H, RG, NRG, H, H, 0, 1);

        // GIN on reduced graph
        zero_f32<<<2048, 256, 0, stream>>>(AGG, NRG * H);
        rg_agg<<<ERG, 128, 0, stream>>>(rg_edge_idx, RG, AGG, ERG);
        gemm(RG, AGG, nullptr, nullptr, rg_W1 + (size_t)i * H * 2 * H, T, NRG, H, 2 * H, 0, 0);
        bnstats(T, NRG, 2 * H);
        gemm(T, nullptr, MEAN, INV, rg_W2 + (size_t)i * 2 * H * H, B2, NRG, 2 * H, H, 1, 0);
        bnstats(B2, NRG, H);
        norm_relu<<<2048, 256, 0, stream>>>(B2, MEAN, INV, RG, NRG * H, H);

        // reduced-graph -> raw: X += relu(scatter_mean(RG[col]->row) @ rg2raw_W)
        zero_f32<<<2048, 256, 0, stream>>>(T, N * H);
        zero_f32<<<256, 256, 0, stream>>>(CNT, N);
        scat_add<<<Mm, 128, 0, stream>>>(mapping_col, mapping_row, RG, T, CNT, Mm);
        div_cnt<<<N, 128, 0, stream>>>(T, CNT, N);
        gemm(T, nullptr, nullptr, nullptr, rg2raw_W + (size_t)i * H * H, X, N, H, H, 0, 1);
    }

    // readout
    zero_f32<<<512, 256, 0, stream>>>(PX, G * H);
    zero_f32<<<512, 256, 0, stream>>>(PRG, G * H);
    zero_f32<<<64, 256, 0, stream>>>(CNT, G);
    pool_add<<<N, 128, 0, stream>>>(batch, X, PX, CNT, N);
    div_cnt<<<G, 128, 0, stream>>>(PX, CNT, G);
    zero_f32<<<64, 256, 0, stream>>>(CNT, G);
    pool_add<<<NRG, 128, 0, stream>>>(tree_batch, RG, PRG, CNT, NRG);
    div_cnt<<<G, 128, 0, stream>>>(PRG, CNT, G);

    gemm(PX,  nullptr, nullptr, nullptr, atom_lin_W, Z, G, H, H, 0, 0);
    gemm(PRG, nullptr, nullptr, nullptr, rg_lin_W,   Z, G, H, H, 0, 2);
    gemm(Z,   nullptr, nullptr, nullptr, lin_W, (float*)d_out, G, H, H, 2, 0);
}
